// HeteroGNN_57174604644451
// MI455X (gfx1250) — matrix operand order repaired and checked by reading, not compiled, not run on hardware
//
#include <hip/hip_runtime.h>
#include <hip/hip_bf16.h>

typedef __attribute__((ext_vector_type(16))) _Float16 v16h;
typedef __attribute__((ext_vector_type(8)))  float    v8f;

#define NEG_SLOPE 0.2f

// ---------------------------------------------------------------------------
// WMMA GEMM: C[N,64] = act(A[N,K] @ W[K,64] + bias), K in {32,64,128}
// f16 inputs (converted on the fly), f32 accumulate via v_wmma_f32_16x16x32_f16.
// One wave -> 16-row strip x 64 cols (4 col tiles). Block = 256 thr = 8 waves.
// W is pre-packed in LDS into per-lane B-fragment order so the inner loop
// loads each 16-half fragment with wide ds_load_b128s (32B contiguous/aligned).
// ---------------------------------------------------------------------------
template <int K>
__global__ void gemm64_wmma(const float* __restrict__ A, const float* __restrict__ W,
                            const float* __restrict__ bias, float* __restrict__ C,
                            int N, int act)
{
    constexpr int KB = K / 32;                 // k-blocks of 32
    __shared__ __align__(32) _Float16 Bp[K * 64];

    // Pack W[k,c] -> Bp[((kb*4 + t)*32 + lane)*16 + i]
    // where col = t*16 + (lane&15), k = kb*32 + (i>>3)*16 + (lane>>4)*8 + (i&7)   (ISA 7.12.2 B layout)
    for (int j = threadIdx.x; j < K * 64; j += blockDim.x) {
        int i    = j & 15;
        int lane = (j >> 4) & 31;
        int t    = (j >> 9) & 3;
        int kb   = j >> 11;
        int col  = t * 16 + (lane & 15);
        int k    = kb * 32 + ((i >> 3) << 4) + ((lane >> 4) << 3) + (i & 7);
        Bp[j] = (_Float16)W[k * 64 + col];
    }
    __syncthreads();

    int waveInBlk = threadIdx.x >> 5;
    int lane      = threadIdx.x & 31;
    int wave      = blockIdx.x * 8 + waveInBlk;
    int rowBase   = wave * 16;
    if (rowBase >= N) return;

    int hi  = lane >> 4;   // 0 or 1 (lane half)
    int l15 = lane & 15;

    int rowA = rowBase + l15;
    if (rowA >= N) rowA = N - 1;          // clamp (stores are guarded)
    const float* Arow = A + (size_t)rowA * K;

    // speculative prefetch of the next block's corresponding A row strip
    __builtin_prefetch(Arow + (size_t)128 * K, 0, 1);

    v8f acc[4];
    acc[0] = (v8f){}; acc[1] = (v8f){}; acc[2] = (v8f){}; acc[3] = (v8f){};

#pragma unroll
    for (int kb = 0; kb < KB; ++kb) {
        // A fragment (16x32 f16): lanes<16 hold K {0..7,16..23}, lanes>=16 {8..15,24..31}
        int kA = kb * 32 + hi * 8;
        float4 q0 = *(const float4*)(Arow + kA);
        float4 q1 = *(const float4*)(Arow + kA + 4);
        float4 q2 = *(const float4*)(Arow + kA + 16);
        float4 q3 = *(const float4*)(Arow + kA + 20);
        v16h a;
        a[0]=(_Float16)q0.x;  a[1]=(_Float16)q0.y;  a[2]=(_Float16)q0.z;  a[3]=(_Float16)q0.w;
        a[4]=(_Float16)q1.x;  a[5]=(_Float16)q1.y;  a[6]=(_Float16)q1.z;  a[7]=(_Float16)q1.w;
        a[8]=(_Float16)q2.x;  a[9]=(_Float16)q2.y;  a[10]=(_Float16)q2.z; a[11]=(_Float16)q2.w;
        a[12]=(_Float16)q3.x; a[13]=(_Float16)q3.y; a[14]=(_Float16)q3.z; a[15]=(_Float16)q3.w;

#pragma unroll
        for (int t = 0; t < 4; ++t) {
            v16h b = *(const v16h*)(Bp + (((kb * 4 + t) * 32 + lane) << 4));
            acc[t] = __builtin_amdgcn_wmma_f32_16x16x32_f16(
                false, a, false, b, (short)0, acc[t], false, false);
        }
    }

    // Epilogue: C/D layout: row = rowBase + r + 8*hi, col = t*16 + (lane&15)
    float* Cw = C + (size_t)(rowBase + hi * 8) * 64 + l15;
    if (rowBase + 16 <= N) {
        // full tile: unguarded stores (half-wave coalesced b32 per row)
#pragma unroll
        for (int t = 0; t < 4; ++t) {
            float bv = bias ? bias[t * 16 + l15] : 0.0f;
#pragma unroll
            for (int r = 0; r < 8; ++r) {
                float v = acc[t][r] + bv;
                if (act == 1) v = fmaxf(v, 0.0f);
                Cw[(size_t)r * 64 + t * 16] = v;
            }
        }
    } else {
#pragma unroll
        for (int t = 0; t < 4; ++t) {
            float bv = bias ? bias[t * 16 + l15] : 0.0f;
#pragma unroll
            for (int r = 0; r < 8; ++r) {
                int row = rowBase + r + hi * 8;
                if (row < N) {
                    float v = acc[t][r] + bv;
                    if (act == 1) v = fmaxf(v, 0.0f);
                    Cw[(size_t)r * 64 + t * 16] = v;
                }
            }
        }
    }
}

// ---------------------------------------------------------------------------
// Small helpers
// ---------------------------------------------------------------------------
__global__ void fill_zero4(float4* p, long long n4)
{
    long long i = (long long)blockIdx.x * blockDim.x + threadIdx.x;
    long long stride = (long long)gridDim.x * blockDim.x;
    float4 z = {0.0f, 0.0f, 0.0f, 0.0f};
    for (; i < n4; i += stride) p[i] = z;
}

// was[k] = sum_c W[k,c]*a_src[c]; wad[k] likewise (alpha_src = h_src . was)
__global__ void prep_wa(const float* __restrict__ W, const float* __restrict__ as,
                        const float* __restrict__ ad, float* __restrict__ wa)
{
    int t = threadIdx.x;              // 128 threads
    const float* av = (t < 64) ? as : ad;
    int k = t & 63;
    float s = 0.0f;
    for (int c = 0; c < 64; ++c) s += W[k * 64 + c] * av[c];
    wa[t] = s;
}

// alpha[n] = leaky_relu(h_src[n,:].was + h_dst[n,:].wad)
__global__ void alpha_kernel(const float* __restrict__ hs, const float* __restrict__ hd,
                             const float* __restrict__ wa, float* __restrict__ alpha, int N)
{
    int n = blockIdx.x * blockDim.x + threadIdx.x;
    if (n >= N) return;
    const float* a = hs + (size_t)n * 64;
    const float* b = hd + (size_t)n * 64;
    float acc = 0.0f;
#pragma unroll
    for (int i = 0; i < 64; ++i) acc += a[i] * wa[i] + b[i] * wa[64 + i];
    alpha[n] = acc > 0.0f ? acc : NEG_SLOPE * acc;
}

__device__ __forceinline__ unsigned enc_f(float f)
{
    unsigned u = __float_as_uint(f);
    return (u & 0x80000000u) ? ~u : (u | 0x80000000u);
}
__device__ __forceinline__ float dec_f(unsigned k)
{
    unsigned u = (k & 0x80000000u) ? (k ^ 0x80000000u) : ~k;
    return __uint_as_float(u);
}

__global__ void init_red(unsigned* mk, float* denom)
{
    *mk = enc_f(-3.402823466e38f);
    *denom = 0.0f;
}

__global__ void edge_max(const float* __restrict__ alpha, const int* __restrict__ s,
                         const int* __restrict__ d, long long E, unsigned* mk)
{
    __shared__ float sm[256];
    float m = -3.402823466e38f;
    long long stride = (long long)gridDim.x * blockDim.x;
    for (long long i = (long long)blockIdx.x * blockDim.x + threadIdx.x; i < E; i += stride)
        m = fmaxf(m, alpha[s[i]] + alpha[d[i]]);
    sm[threadIdx.x] = m; __syncthreads();
    for (int o = 128; o > 0; o >>= 1) {
        if ((int)threadIdx.x < o) sm[threadIdx.x] = fmaxf(sm[threadIdx.x], sm[threadIdx.x + o]);
        __syncthreads();
    }
    if (threadIdx.x == 0) atomicMax(mk, enc_f(sm[0]));
}

__global__ void edge_sumexp(const float* __restrict__ alpha, const int* __restrict__ s,
                            const int* __restrict__ d, long long E,
                            const unsigned* mk, float* denom)
{
    __shared__ float sm[256];
    float M = dec_f(*mk);
    float acc = 0.0f;
    long long stride = (long long)gridDim.x * blockDim.x;
    for (long long i = (long long)blockIdx.x * blockDim.x + threadIdx.x; i < E; i += stride)
        acc += __expf((alpha[s[i]] + alpha[d[i]]) - M);
    sm[threadIdx.x] = acc; __syncthreads();
    for (int o = 128; o > 0; o >>= 1) {
        if ((int)threadIdx.x < o) sm[threadIdx.x] += sm[threadIdx.x + o];
        __syncthreads();
    }
    if (threadIdx.x == 0) atomicAdd(denom, sm[0]);
}

// agg[d_e,:] += xs[s_e,:] * softmax_e ; 16 lanes per edge, float4 per lane
__global__ void edge_scatter(const float* __restrict__ alpha, const int* __restrict__ sidx,
                             const int* __restrict__ didx, long long E,
                             const unsigned* mk, const float* denom,
                             const float* __restrict__ xs, float* __restrict__ agg)
{
    long long t = (long long)blockIdx.x * blockDim.x + threadIdx.x;
    long long e = t >> 4;
    if (e >= E) return;
    int sub = (int)(t & 15);
    int s = sidx[e], d = didx[e];
    float ae = __expf((alpha[s] + alpha[d]) - dec_f(*mk)) / *denom;
    float4 v = *(const float4*)(xs + (size_t)s * 64 + sub * 4);
    float* o = agg + (size_t)d * 64 + sub * 4;
    atomicAdd(o + 0, v.x * ae);
    atomicAdd(o + 1, v.y * ae);
    atomicAdd(o + 2, v.z * ae);
    atomicAdd(o + 3, v.w * ae);
}

// out = f(agg + b0 + b1 + b2); f = relu(v)+v if residual else v
__global__ void post_layer(const float* __restrict__ agg,
                           const float* __restrict__ b0, const float* __restrict__ b1,
                           const float* __restrict__ b2,
                           float* __restrict__ out, int N, int residual)
{
    long long i = (long long)blockIdx.x * blockDim.x + threadIdx.x;
    long long total = (long long)N * 64;
    if (i >= total) return;
    int c = (int)(i & 63);
    float bs = b0[c] + (b1 ? b1[c] : 0.0f) + (b2 ? b2[c] : 0.0f);
    float v = agg[i] + bs;
    out[i] = residual ? (fmaxf(v, 0.0f) + v) : v;
}

__global__ void colsum(const float* __restrict__ x, long long total, float* __restrict__ sums)
{
    long long tid = (long long)blockIdx.x * blockDim.x + threadIdx.x;
    long long stride = (long long)gridDim.x * blockDim.x;   // multiple of 64
    int col = (int)(tid & 63);
    float acc = 0.0f;
    for (long long i = tid; i < total; i += stride) acc += x[i];
    atomicAdd(sums + col, acc);
}

// pooled = (relu(mean @ W1 + b1)) @ W2 + b2 ; mean = sums/N ; 64 threads, 1 block
__global__ void pool_mlp(const float* __restrict__ sums, float invN,
                         const float* __restrict__ W1, const float* __restrict__ b1,
                         const float* __restrict__ W2, const float* __restrict__ b2,
                         float* __restrict__ out)
{
    __shared__ float mean[64];
    __shared__ float hid[32];
    int t = threadIdx.x;
    mean[t] = sums[t] * invN;
    __syncthreads();
    if (t < 32) {
        float s = b1[t];
        for (int i = 0; i < 64; ++i) s += mean[i] * W1[i * 32 + t];
        hid[t] = fmaxf(s, 0.0f);
    }
    __syncthreads();
    if (t < 16) {
        float s = b2[t];
        for (int i = 0; i < 32; ++i) s += hid[i] * W2[i * 16 + t];
        out[t] = s;
    }
}

// ---------------------------------------------------------------------------
// Host orchestration
// ---------------------------------------------------------------------------
static inline void launch_gemm(const float* A, const float* W, const float* bias, float* C,
                               int N, int K, int act, hipStream_t st)
{
    int rowTiles = (N + 15) / 16;
    int blocks = (rowTiles + 7) / 8;
    if (K == 32)
        gemm64_wmma<32><<<blocks, 256, 0, st>>>(A, W, bias, C, N, act);
    else if (K == 64)
        gemm64_wmma<64><<<blocks, 256, 0, st>>>(A, W, bias, C, N, act);
    else
        gemm64_wmma<128><<<blocks, 256, 0, st>>>(A, W, bias, C, N, act);
}

extern "C" void kernel_launch(void* const* d_in, const int* in_sizes, int n_in,
                              void* d_out, int out_size, void* d_ws, size_t ws_size,
                              hipStream_t stream)
{
    (void)n_in; (void)out_size; (void)ws_size;
    const int N = in_sizes[0] / 32;              // x_patient is [N,32]
    const long long E = in_sizes[3] / 2;         // edges are [2,E]

    // ---- param index map (params pytree: dicts sorted by key) ----
    // 0..2  x_patient,x_stay,x_day ; 3..8 edges (p2s, s2d, has_lab, has_vital, rev_p2s, rev_s2d)
    // 9..14 in_proj: day(W,b), patient(W,b), stay(W,b)
    // 15..38 layer0, 39..62 layer1: per et sorted {has_lab,has_vital,p2s,rev_p2s,rev_s2d,s2d} x (W,a_src,a_dst,b)
    // 63..74 out_proj: day(W1,b1,W2,b2), patient(...), stay(...)
    // 75..86 pool:     day(W1,b1,W2,b2), patient(...), stay(...)
#define FP(i) ((const float*)d_in[(i)])
#define IP(i) ((const int*)d_in[(i)])

    // workspace layout
    char* w = (char*)d_ws;
    const size_t S = (size_t)N * 64 * sizeof(float);
    float* h[3];   h[0] = (float*)w; w += S; h[1] = (float*)w; w += S; h[2] = (float*)w; w += S;
    float* agg[3]; agg[0] = (float*)w; w += S; agg[1] = (float*)w; w += S; agg[2] = (float*)w; w += S;
    float* xsb   = (float*)w; w += S;
    float* alpha = (float*)w; w += (size_t)N * sizeof(float);
    float* wa    = (float*)w; w += 128 * sizeof(float);
    unsigned* mk = (unsigned*)w; w += 64;       // keep 64B aligned
    float* denom = (float*)w; w += 64;
    float* csum  = (float*)w; w += 64 * sizeof(float);

    float* out = (float*)d_out;

    // ---- in_proj ----
    launch_gemm(FP(0), FP(11), FP(12), h[0], N, 32, 0, stream);   // patient
    launch_gemm(FP(1), FP(13), FP(14), h[1], N, 64, 0, stream);   // stay
    launch_gemm(FP(2), FP(9),  FP(10), h[2], N, 128, 0, stream);  // day

    // relation table: {edge input idx, sorted param slot, src nt, dst nt}
    const int rel_ei[6]  = {3, 4, 5, 6, 7, 8};
    const int rel_ps[6]  = {2, 5, 0, 1, 3, 4};   // p2s,s2d,has_lab,has_vital,rev_p2s,rev_s2d
    const int rel_st[6]  = {0, 1, 2, 2, 1, 2};
    const int rel_dt[6]  = {1, 2, 2, 2, 0, 1};

    const int nAlphaBlk   = (N + 255) / 256;
    const int nRedBlk     = (int)(((E + 255) / 256) < 1024 ? ((E + 255) / 256) : 1024);
    const int nScatterBlk = (int)((E * 16 + 255) / 256);
    const long long totalNC = (long long)N * 64;
    const int nElemBlk    = (int)((totalNC + 255) / 256);

    // ---- GAT layers ----
    for (int L = 0; L < 2; ++L) {
        const int base = 15 + L * 24;
        fill_zero4<<<1024, 256, 0, stream>>>((float4*)agg[0], 3LL * totalNC / 4);

        for (int r = 0; r < 6; ++r) {
            const int pw = base + rel_ps[r] * 4;   // W, a_src, a_dst, b
            const int st = rel_st[r], dt = rel_dt[r];
            const int* eb = IP(rel_ei[r]);
            const int* sI = eb;
            const int* dI = eb + E;

            prep_wa<<<1, 128, 0, stream>>>(FP(pw), FP(pw + 1), FP(pw + 2), wa);
            launch_gemm(h[st], FP(pw), nullptr, xsb, N, 64, 0, stream);
            alpha_kernel<<<nAlphaBlk, 256, 0, stream>>>(h[st], h[dt], wa, alpha, N);
            init_red<<<1, 1, 0, stream>>>(mk, denom);
            edge_max<<<nRedBlk, 256, 0, stream>>>(alpha, sI, dI, E, mk);
            edge_sumexp<<<nRedBlk, 256, 0, stream>>>(alpha, sI, dI, E, mk, denom);
            edge_scatter<<<nScatterBlk, 256, 0, stream>>>(alpha, sI, dI, E, mk, denom, xsb, agg[dt]);
        }

        // per-destination bias sums (relations targeting each node type)
        // patient <- rev_p2s(slot3); stay <- p2s(slot2), rev_s2d(slot4); day <- s2d(slot5), has_lab(0), has_vital(1)
        const float* pb[3][3] = {
            { FP(base + 3 * 4 + 3), nullptr,              nullptr              },
            { FP(base + 2 * 4 + 3), FP(base + 4 * 4 + 3), nullptr              },
            { FP(base + 5 * 4 + 3), FP(base + 0 * 4 + 3), FP(base + 1 * 4 + 3) },
        };
        const int residual = (L == 0) ? 1 : 0;
        for (int nt = 0; nt < 3; ++nt)
            post_layer<<<nElemBlk, 256, 0, stream>>>(agg[nt], pb[nt][0], pb[nt][1], pb[nt][2],
                                                     h[nt], N, residual);
    }

    // ---- out_proj + pooling ----
    const int opIdx[3]   = {67, 71, 63};   // patient, stay, day
    const int poolIdx[3] = {79, 83, 75};
    for (int nt = 0; nt < 3; ++nt) {
        float* proc = out + (size_t)nt * totalNC;
        launch_gemm(h[nt], FP(opIdx[nt]),     FP(opIdx[nt] + 1), xsb,  N, 64, 1, stream);
        launch_gemm(xsb,   FP(opIdx[nt] + 2), FP(opIdx[nt] + 3), proc, N, 64, 0, stream);

        fill_zero4<<<1, 64, 0, stream>>>((float4*)csum, 16);
        colsum<<<256, 256, 0, stream>>>(proc, totalNC, csum);
        pool_mlp<<<1, 64, 0, stream>>>(csum, 1.0f / (float)N,
                                       FP(poolIdx[nt]), FP(poolIdx[nt] + 1),
                                       FP(poolIdx[nt] + 2), FP(poolIdx[nt] + 3),
                                       out + 3 * totalNC + nt * 16);
    }
#undef FP
#undef IP
}
